// EdgeNetwork_70712341561618
// MI455X (gfx1250) — compile-verified, matrix operand order checked by reading, not run on hardware
//
#include <hip/hip_runtime.h>
#include <hip/hip_bf16.h>

#define N_NODES 50000
#define N_EDGES 150000
#define EMB     64
#define IN_DIM  32
#define LN_EPS  1e-5f
#define TILE_E  64       // edges per block (N-tile)
#define K_CHUNKS 66      // 64 chunks of W (c = 0..63) + 2 bias chunks
#define CHUNK_ELEMS (EMB * IN_DIM)          // 2048 bf16 per chunk
#define APACK_ELEMS (K_CHUNKS * CHUNK_ELEMS)

typedef __attribute__((ext_vector_type(16))) __bf16 v16bf;
typedef __attribute__((ext_vector_type(8)))  __bf16 v8bf;
typedef __attribute__((ext_vector_type(8)))  float  v8f;
typedef __attribute__((ext_vector_type(4)))  unsigned int v4u;
typedef __attribute__((ext_vector_type(8)))  int v8i;
typedef __attribute__((ext_vector_type(4)))  int v4i;

static __device__ inline v16bf combine16(v8bf lo, v8bf hi) {
    return __builtin_shufflevector(lo, hi, 0,1,2,3,4,5,6,7,8,9,10,11,12,13,14,15);
}

// Issue a TDM transfer of one contiguous 4KB A-chunk (2048 bf16) into LDS.
// D# built per CDNA5 ISA ch.8: 1-D tile, data_size=2B, tensor_dim0=tile_dim0=2048.
static __device__ inline void tdm_load_chunk(const __bf16* gsrc, unsigned lds_off) {
    unsigned long long ga = (unsigned long long)(uintptr_t)gsrc;
    v4u g0;
    g0[0] = 1u;                                        // count=1, user descriptor
    g0[1] = lds_off;                                   // lds_addr [63:32]
    g0[2] = (unsigned)(ga & 0xFFFFFFFFu);              // global_addr lo
    g0[3] = (unsigned)((ga >> 32) & 0x01FFFFFFu)       // global_addr hi [120:96]
          | (2u << 30);                                // type=2 ("image")
    v8i g1;
    g1[0] = 0x00010000;                                // data_size=1 (2 bytes)
    g1[1] = (int)((unsigned)CHUNK_ELEMS << 16);        // tensor_dim0[15:0] @ [63:48]
    g1[2] = (int)(1u << 16);                           // tensor_dim0 hi=0; tensor_dim1=1
    g1[3] = (int)((unsigned)CHUNK_ELEMS << 16);        // tensor_dim1 hi=0; tile_dim0=2048
    g1[4] = 0;                                         // tile_dim1=0, tile_dim2=0 (1-D)
    g1[5] = CHUNK_ELEMS;                               // tensor_dim0_stride lo
    g1[6] = 0;
    g1[7] = 0;
    v4i z4 = {0, 0, 0, 0};
    v8i z8 = {0, 0, 0, 0, 0, 0, 0, 0};
    // 6-arg form (clang-23 / therock-10.0 headers): groups 0..3 + aux + cpol
    __builtin_amdgcn_tensor_load_to_lds(g0, g1, z4, z4, z8, 0);
}

// ---------------------------------------------------------------------------
// Kernel P: pack [Wr | B64] chunk-major as bf16 into workspace.
//   Apack[cc][r][k]: cc<64 -> W[(r*64+cc)*32+k]; cc>=64 -> b[r*64+(cc-64)*32+k]
// ---------------------------------------------------------------------------
__global__ __launch_bounds__(256)
void edgenet_pack_kernel(const float* __restrict__ W, const float* __restrict__ b,
                         __bf16* __restrict__ Apack) {
    int idx = blockIdx.x * blockDim.x + threadIdx.x;
    if (idx >= APACK_ELEMS) return;
    int cc = idx >> 11;          // /2048
    int r  = (idx >> 5) & 63;
    int k  = idx & 31;
    float w;
    if (cc < 64) w = W[((r << 6) + cc) * IN_DIM + k];
    else         w = b[(r << 6) + ((cc - 64) << 5) + k];
    Apack[idx] = (__bf16)w;
}

// ---------------------------------------------------------------------------
// Kernel 0: out = h  (residual baseline; atomics accumulate on top)
// ---------------------------------------------------------------------------
__global__ __launch_bounds__(256)
void edgenet_copy_kernel(const float* __restrict__ src, float* __restrict__ dst, int n4) {
    int i = blockIdx.x * blockDim.x + threadIdx.x;
    if (i < n4) ((float4*)dst)[i] = ((const float4*)src)[i];
}

// ---------------------------------------------------------------------------
// Kernel 1: per-edge message GEMM via bf16 WMMA; A chunks streamed by TDM
// (double-buffered, s_wait_tensorcnt pipelined); fused atomic scatter.
// ---------------------------------------------------------------------------
__global__ __launch_bounds__(256)
void edgenet_msg_kernel(const float* __restrict__ h,
                        const float* __restrict__ ea,
                        const __bf16* __restrict__ Apack,
                        const int*  __restrict__ e_src,   // scatter target (row 0)
                        const int*  __restrict__ e_dst,   // gather source (row 1)
                        float* __restrict__ out) {
    __shared__ __attribute__((aligned(128))) __bf16 sA[2][EMB][IN_DIM]; // 2x4KB
    __shared__ float  sX[TILE_E][EMB];        // gathered node features (16 KB)
    __shared__ float  sAttr[TILE_E][IN_DIM];  // edge attrs              (8 KB)
    __shared__ int    sSrc[TILE_E];
    __shared__ int    sDstN[TILE_E];

    const int t      = threadIdx.x;
    const int lane   = t & 31;
    const int wave   = t >> 5;                 // 0..7
    const int m_tile = wave & 3;               // M strip: rows m_tile*16..+15
    const int n0     = (wave >> 2) * 2;        // two N tiles: n0, n0+1
    const int half   = lane >> 4;              // 0/1
    const int mn     = lane & 15;              // row within M tile / col within N tile
    const int e_base = blockIdx.x * TILE_E;
    const bool full  = (e_base + TILE_E) <= N_EDGES;

    // kick off the first TDM transfer ASAP (overlaps with gather prologue)
    if (wave == 0) {
        tdm_load_chunk(Apack, (unsigned)(uintptr_t)&sA[0][0][0]);
    }

    // ---- stage edge indices ----
    if (t < TILE_E) {
        int e = e_base + t;
        sSrc[t]  = (e < N_EDGES) ? e_src[e] : 0;
        sDstN[t] = (e < N_EDGES) ? e_dst[e] : 0;
    }
    __syncthreads();

    // ---- stage gathered node features: 4 threads/row, float4 moves ----
    {
        int e = t >> 2, part = t & 3;          // 64 rows x 4 threads
        float4* drow = (float4*)&sX[e][0];
        if (full || (e_base + e) < N_EDGES) {
            const float4* srow = (const float4*)(h + (long long)sDstN[e] * EMB);
            #pragma unroll
            for (int q = 0; q < 4; ++q) drow[part * 4 + q] = srow[part * 4 + q];
        } else {
            float4 z = {0.f, 0.f, 0.f, 0.f};
            #pragma unroll
            for (int q = 0; q < 4; ++q) drow[part * 4 + q] = z;
        }
    }
    // ---- stage edge attributes: 4 threads/row, float4 moves ----
    {
        int e = t >> 2, part = t & 3;
        float4* drow = (float4*)&sAttr[e][0];
        if (full || (e_base + e) < N_EDGES) {
            const float4* srow = (const float4*)(ea + (long long)(e_base + e) * IN_DIM);
            #pragma unroll
            for (int q = 0; q < 2; ++q) drow[part * 2 + q] = srow[part * 2 + q];
        } else {
            float4 z = {0.f, 0.f, 0.f, 0.f};
            #pragma unroll
            for (int q = 0; q < 2; ++q) drow[part * 2 + q] = z;
        }
    }

    v8f acc0 = {};
    v8f acc1 = {};
    const int e0 = n0 * 16 + mn;
    const int e1 = (n0 + 1) * 16 + mn;

    for (int cc = 0; cc < K_CHUNKS; ++cc) {
        const int buf = cc & 1;
        // prefetch next chunk via TDM, then wait for current chunk's DMA
        if (wave == 0) {
            if (cc + 1 < K_CHUNKS) {
                tdm_load_chunk(Apack + (long long)(cc + 1) * CHUNK_ELEMS,
                               (unsigned)(uintptr_t)&sA[buf ^ 1][0][0]);
                __builtin_amdgcn_s_wait_tensorcnt(1);
            } else {
                __builtin_amdgcn_s_wait_tensorcnt(0);
            }
        }
        __syncthreads();   // sA[buf] ready for everyone; prologue also fenced

        // ---- A fragment (16x32 bf16, ISA layout) ----
        const int r = m_tile * 16 + mn;
        v8bf alo = *(const v8bf*)&sA[buf][r][half * 8];
        v8bf ahi = *(const v8bf*)&sA[buf][r][16 + half * 8];
        v16bf afrag = combine16(alo, ahi);

        // ---- B fragments synthesized from outer product x (X) a ----
        v16bf bfrag0, bfrag1;
        if (cc < 64) {
            float xc0 = sX[e0][cc];
            float xc1 = sX[e1][cc];
            #pragma unroll
            for (int j = 0; j < 8; ++j) {
                int klo = half * 8 + j;
                int khi = 16 + half * 8 + j;
                bfrag0[j]     = (__bf16)(sAttr[e0][klo] * xc0);
                bfrag0[j + 8] = (__bf16)(sAttr[e0][khi] * xc0);
                bfrag1[j]     = (__bf16)(sAttr[e1][klo] * xc1);
                bfrag1[j + 8] = (__bf16)(sAttr[e1][khi] * xc1);
            }
        } else {
            int base = (cc - 64) * 32;
            #pragma unroll
            for (int j = 0; j < 8; ++j) {
                int klo = base + half * 8 + j;
                int khi = base + 16 + half * 8 + j;
                bfrag0[j]     = (__bf16)sX[e0][klo];
                bfrag0[j + 8] = (__bf16)sX[e0][khi];
                bfrag1[j]     = (__bf16)sX[e1][klo];
                bfrag1[j + 8] = (__bf16)sX[e1][khi];
            }
        }

        acc0 = __builtin_amdgcn_wmma_f32_16x16x32_bf16(
                   false, afrag, false, bfrag0, (short)0, acc0, false, false);
        acc1 = __builtin_amdgcn_wmma_f32_16x16x32_bf16(
                   false, afrag, false, bfrag1, (short)0, acc1, false, false);
        __syncthreads();   // all waves done with sA[buf] before TDM reuses it
    }

    // ---- fused scatter-add: C/D layout M = v + 8*half, N = lane%16 ----
    {
        int e = e_base + e0;
        if (e < N_EDGES) {
            long long nb = (long long)sSrc[e0] * EMB;
            #pragma unroll
            for (int v = 0; v < 8; ++v) {
                int rr = m_tile * 16 + v + 8 * half;
                atomicAdd(&out[nb + rr], acc0[v]);
            }
        }
        e = e_base + e1;
        if (e < N_EDGES) {
            long long nb = (long long)sSrc[e1] * EMB;
            #pragma unroll
            for (int v = 0; v < 8; ++v) {
                int rr = m_tile * 16 + v + 8 * half;
                atomicAdd(&out[nb + rr], acc1[v]);
            }
        }
    }
}

// ---------------------------------------------------------------------------
// Kernel 2: in-place residual LayerNorm, one wave32 per node (2 elems/lane)
// ---------------------------------------------------------------------------
__global__ __launch_bounds__(256)
void edgenet_ln_kernel(float* __restrict__ out,
                       const float* __restrict__ gamma,
                       const float* __restrict__ beta) {
    int wave = threadIdx.x >> 5;
    int lane = threadIdx.x & 31;
    int node = blockIdx.x * 8 + wave;
    if (node >= N_NODES) return;

    float2 v = *(const float2*)&out[(long long)node * EMB + lane * 2];
    float s = v.x + v.y;
    #pragma unroll
    for (int m = 16; m > 0; m >>= 1) s += __shfl_xor(s, m, 32);
    float mu = s * (1.0f / EMB);
    float dx = v.x - mu, dy = v.y - mu;
    float q = dx * dx + dy * dy;
    #pragma unroll
    for (int m = 16; m > 0; m >>= 1) q += __shfl_xor(q, m, 32);
    float inv = rsqrtf(q * (1.0f / EMB) + LN_EPS);

    float2 g  = *(const float2*)&gamma[lane * 2];
    float2 be = *(const float2*)&beta[lane * 2];
    float2 o;
    o.x = dx * inv * g.x + be.x;
    o.y = dy * inv * g.y + be.y;
    *(float2*)&out[(long long)node * EMB + lane * 2] = o;
}

// ---------------------------------------------------------------------------
extern "C" void kernel_launch(void* const* d_in, const int* in_sizes, int n_in,
                              void* d_out, int out_size, void* d_ws, size_t ws_size,
                              hipStream_t stream) {
    const float* h     = (const float*)d_in[0];
    const float* ea    = (const float*)d_in[1];
    const float* W     = (const float*)d_in[2];
    const float* b     = (const float*)d_in[3];
    const float* gamma = (const float*)d_in[4];
    const float* beta  = (const float*)d_in[5];
    const int*   ei    = (const int*)d_in[6];   // [2, N_EDGES]: row0=src, row1=dst
    float* out = (float*)d_out;
    __bf16* Apack = (__bf16*)d_ws;              // 270 KB, L2-resident

    (void)in_sizes; (void)n_in; (void)out_size; (void)ws_size;

    // pack [Wr | B64] as bf16, chunk-major (TDM source)
    edgenet_pack_kernel<<<(APACK_ELEMS + 255) / 256, 256, 0, stream>>>(W, b, Apack);

    // out = h (residual base for atomic accumulation)
    int n4 = N_NODES * EMB / 4;
    edgenet_copy_kernel<<<(n4 + 255) / 256, 256, 0, stream>>>(h, out, n4);

    // per-edge WMMA GEMM + fused scatter, TDM-fed A chunks
    int eblocks = (N_EDGES + TILE_E - 1) / TILE_E;
    edgenet_msg_kernel<<<eblocks, 256, 0, stream>>>(h, ea, Apack,
                                                    ei, ei + N_EDGES, out);

    // in-place LayerNorm
    int lblocks = (N_NODES + 7) / 8;
    edgenet_ln_kernel<<<lblocks, 256, 0, stream>>>(out, gamma, beta);
}
